// AttentionModule_3255585210804
// MI455X (gfx1250) — matrix operand order repaired and checked by reading, not compiled, not run on hardware
//
#include <hip/hip_runtime.h>

// ---------------------------------------------------------------------------
// CDNA5 / gfx1250 (MI455X) relation-network attention.
//
// Algebra:  out[n, g*64+o] = sum_m P[n,g,m] * VgT[g*64+o, m] + conv_b,
//           VgT = conv_w @ roi[:512]^T   (kills the 67MB out_t intermediate,
//           19.3 GFLOP -> 3.2 GFLOP).
//
// All WMMA operands pre-packed to f16 (pack pass + f16 q/k/VgT/logw), so
// GEMM & flash inner loops are pure load -> v_wmma.  Softmax is carried in
// the log2 domain (exact: softmax is invariant under x -> x*log2e with
// exp -> exp2), so transcendentals are raw v_log_f32 / v_exp_f32 with no
// denormal-guard fixup code.  Roofline: the one-shot 134MB non-temporal
// position_embedding stream dominates -> ~6us at 23.3 TB/s HBM.
// ---------------------------------------------------------------------------

typedef __attribute__((ext_vector_type(16))) _Float16 v16h;
typedef __attribute__((ext_vector_type(8)))  _Float16 h8;
typedef __attribute__((ext_vector_type(8)))  float    v8f;
typedef __attribute__((ext_vector_type(4)))  float    f4;

__device__ __forceinline__ v8f wmma_f16(v16h a, v16h b, v8f c) {
  return __builtin_amdgcn_wmma_f32_16x16x32_f16(
      /*neg_a=*/false, a, /*neg_b=*/false, b,
      /*c_mod=*/(short)0, c, /*reuse_a=*/false, /*reuse_b=*/false);
}

// Raw hardware base-2 transcendentals (inputs are never denormal here).
__device__ __forceinline__ float fast_log2(float x) {
#if __has_builtin(__builtin_amdgcn_logf)
  return __builtin_amdgcn_logf(x);
#else
  return __log2f(x);
#endif
}
__device__ __forceinline__ float fast_exp2(float x) {
#if __has_builtin(__builtin_amdgcn_exp2f)
  return __builtin_amdgcn_exp2f(x);
#else
  return exp2f(x);
#endif
}

// ---------------- f32 tile loaders (only for the one-shot pe stream) -------
// A-matrix (16 rows x 32 K): lane l<16 -> row0+l, K {0..7,16..23};
// lane l+16 -> K {8..15,24..31}.  Non-temporal: stream, don't pollute L2.
__device__ __forceinline__ v16h load_tileA_f32_nt(const float* __restrict__ src,
                                                  int ld, int row0, int k0, int lane) {
  const int r = lane & 15, hs = lane >> 4;
  const float* p = src + (size_t)(row0 + r) * ld + k0 + hs * 8;
  const f4 x0 = __builtin_nontemporal_load((const f4*)(p));
  const f4 x1 = __builtin_nontemporal_load((const f4*)(p + 4));
  const f4 x2 = __builtin_nontemporal_load((const f4*)(p + 16));
  const f4 x3 = __builtin_nontemporal_load((const f4*)(p + 20));
  v16h a;
#pragma unroll
  for (int i = 0; i < 4; ++i) {
    a[i]      = (_Float16)x0[i];
    a[i + 4]  = (_Float16)x1[i];
    a[i + 8]  = (_Float16)x2[i];
    a[i + 12] = (_Float16)x3[i];
  }
  return a;
}

// B-matrix (32 K x 16 cols), B[k][n] = Bmat[n0+n][k0+k], Bmat row-major [N,ld].
// Lane l: col n = l&15, K = (l>>4)*8 + {0..7,16..23}.
__device__ __forceinline__ v16h load_tileB_f32(const float* __restrict__ src,
                                               int ld, int n0, int k0, int lane) {
  const int c = lane & 15, hs = lane >> 4;
  const float* p = src + (size_t)(n0 + c) * ld + k0 + hs * 8;
  const f4 x0 = *(const f4*)(p);
  const f4 x1 = *(const f4*)(p + 4);
  const f4 x2 = *(const f4*)(p + 16);
  const f4 x3 = *(const f4*)(p + 20);
  v16h b;
#pragma unroll
  for (int i = 0; i < 4; ++i) {
    b[i]      = (_Float16)x0[i];
    b[i + 4]  = (_Float16)x1[i];
    b[i + 8]  = (_Float16)x2[i];
    b[i + 12] = (_Float16)x3[i];
  }
  return b;
}

// ---------------- f16 tile loaders (steady state: 2 x 16B loads, no cvt) ---
__device__ __forceinline__ v16h load_tileA_h(const _Float16* __restrict__ src,
                                             int ld, int row0, int k0, int lane) {
  const int r = lane & 15, hs = lane >> 4;
  const _Float16* p = src + (size_t)(row0 + r) * ld + k0 + hs * 8;
  const h8 lo = *(const h8*)(p);
  const h8 hi = *(const h8*)(p + 16);
  return __builtin_shufflevector(lo, hi, 0, 1, 2, 3, 4, 5, 6, 7,
                                 8, 9, 10, 11, 12, 13, 14, 15);
}

__device__ __forceinline__ v16h load_tileB_h(const _Float16* __restrict__ src,
                                             int ld, int n0, int k0, int lane) {
  const int c = lane & 15, hs = lane >> 4;
  const _Float16* p = src + (size_t)(n0 + c) * ld + k0 + hs * 8;
  const h8 lo = *(const h8*)(p);
  const h8 hi = *(const h8*)(p + 16);
  return __builtin_shufflevector(lo, hi, 0, 1, 2, 3, 4, 5, 6, 7,
                                 8, 9, 10, 11, 12, 13, 14, 15);
}

// ---------------------------------------------------------------------------
// Kernel 0: elementwise f32 -> f16 pack (roi, q_w, k_w, conv_w).  L2 resident.
// ---------------------------------------------------------------------------
__global__ void pack_f16_kernel(const float* __restrict__ in,
                                _Float16* __restrict__ out, int n) {
  const int i = (blockIdx.x * blockDim.x + threadIdx.x) * 8;
  if (i >= n) return;
  const f4 a = *(const f4*)(in + i);
  const f4 b = *(const f4*)(in + i + 4);
  h8 h;
#pragma unroll
  for (int j = 0; j < 4; ++j) {
    h[j]     = (_Float16)a[j];
    h[j + 4] = (_Float16)b[j];
  }
  *(h8*)(out + i) = h;
}

// ---------------------------------------------------------------------------
// Kernel 1: logw[n,g,m] = log2(max(relu(pe[n,m,:].pos_w[g,:] + pos_b[g]),1e-6))
// (log2 domain -- consumed by an exp2-based softmax, mathematically exact.)
// GEMM M = n*512+m (524288), N = 16, K = 64.  HBM-bound (134MB stream, NT).
// Output f16, one packed 16B store per lane (8 consecutive m).
// ---------------------------------------------------------------------------
__global__ void posbias_kernel(const float* __restrict__ pe,
                               const float* __restrict__ pos_w,
                               const float* __restrict__ pos_b,
                               _Float16* __restrict__ logw) {
  const int lane = threadIdx.x & 31;
  const int wave = threadIdx.x >> 5;
  const int row0 = (blockIdx.x * 4 + wave) * 16;  // flattened (n,m)

  const v16h b0 = load_tileB_f32(pos_w, 64, 0, 0, lane);
  const v16h b1 = load_tileB_f32(pos_w, 64, 0, 32, lane);
  const v16h a0 = load_tileA_f32_nt(pe, 64, row0, 0, lane);
  const v16h a1 = load_tileA_f32_nt(pe, 64, row0, 32, lane);

  v8f acc = {};
  acc = wmma_f16(a0, b0, acc);
  acc = wmma_f16(a1, b1, acc);

  const int f  = lane & 15;
  const int hs = lane >> 4;
  const float bb = pos_b[f];

  // rows j=0..7 are 8 consecutive m within one n: pack one 16B store.
  h8 h;
#pragma unroll
  for (int j = 0; j < 8; ++j) {
    float v = acc[j] + bb;
    v = fmaxf(v, 0.0f);     // relu
    v = fmaxf(v, 1e-6f);    // clamp
    h[j] = (_Float16)fast_log2(v);
  }
  const int nm0 = row0 + 8 * hs;
  const int n = nm0 >> 9, m = nm0 & 511;
  *(h8*)(logw + ((size_t)n * 16 + f) * 512 + m) = h;
}

// ---------------------------------------------------------------------------
// Kernel 2 (generic): C[M,N](f16) = A[M,K](f16) @ B[N,K](f16)^T (+bias f32).
// Wave = 16(M) x 64(N) tile, block = 4 waves along M, grid = (N/64, M/64).
// Inner loop: 5 x 32B loads + 4 wmma, zero conversions.
// ---------------------------------------------------------------------------
__global__ void gemm_abt_f16_kernel(const _Float16* __restrict__ A,
                                    const _Float16* __restrict__ B,
                                    const float* __restrict__ bias,
                                    _Float16* __restrict__ C,
                                    int K, int ldc) {
  const int lane = threadIdx.x & 31;
  const int wave = threadIdx.x >> 5;
  const int m0 = blockIdx.y * 64 + wave * 16;
  const int n0 = blockIdx.x * 64;

  v8f acc0 = {}, acc1 = {}, acc2 = {}, acc3 = {};
  for (int k0 = 0; k0 < K; k0 += 32) {
    const v16h a = load_tileA_h(A, K, m0, k0, lane);
    acc0 = wmma_f16(a, load_tileB_h(B, K, n0,      k0, lane), acc0);
    acc1 = wmma_f16(a, load_tileB_h(B, K, n0 + 16, k0, lane), acc1);
    acc2 = wmma_f16(a, load_tileB_h(B, K, n0 + 32, k0, lane), acc2);
    acc3 = wmma_f16(a, load_tileB_h(B, K, n0 + 48, k0, lane), acc3);
  }

  const int c = lane & 15, hs = lane >> 4;
  float b0 = 0.f, b1 = 0.f, b2 = 0.f, b3 = 0.f;
  if (bias) {
    b0 = bias[n0 + c];      b1 = bias[n0 + 16 + c];
    b2 = bias[n0 + 32 + c]; b3 = bias[n0 + 48 + c];
  }
#pragma unroll
  for (int j = 0; j < 8; ++j) {
    const size_t row = (size_t)(m0 + j + 8 * hs);
    C[row * ldc + n0 + c]      = (_Float16)(acc0[j] + b0);
    C[row * ldc + n0 + 16 + c] = (_Float16)(acc1[j] + b1);
    C[row * ldc + n0 + 32 + c] = (_Float16)(acc2[j] + b2);
    C[row * ldc + n0 + 48 + c] = (_Float16)(acc3[j] + b3);
  }
}

// ---------------------------------------------------------------------------
// Kernel 3: flash attention per head, all-f16 operands, log2-domain softmax.
//   St2 = (scale*log2e) * K_g @ Q_g^T + log2w, online exp2-softmax over m,
//   O = P @ VgT^T, epilogue 1/rowsum + conv_b.
// S is computed transposed (n in lanes, m in VGPRs): softmax = per-lane
// reduction + one shfl_xor(16); repack of P into A-layout is lane-local.
// grid = (16 heads, 16), block = 128 (4 waves x 16 query rows).
// ---------------------------------------------------------------------------
__global__ void flash_kernel(const _Float16* __restrict__ q,
                             const _Float16* __restrict__ k,
                             const _Float16* __restrict__ vgt,
                             const _Float16* __restrict__ logw,
                             const float* __restrict__ conv_b,
                             float* __restrict__ out) {
  const int lane = threadIdx.x & 31;
  const int wave = threadIdx.x >> 5;
  const int g  = blockIdx.x;                    // head
  const int n0 = blockIdx.y * 64 + wave * 16;   // query rows
  const int c  = lane & 15, hs = lane >> 4;
  // (1/sqrt(64)) * log2(e): logits scaled into log2 domain.
  const float kScale = 0.125f * 1.44269504088896340736f;

  // Loop-invariant B-tiles of Q^T.
  const v16h bq0 = load_tileB_h(q, 1024, n0, g * 64,      lane);
  const v16h bq1 = load_tileB_h(q, 1024, n0, g * 64 + 32, lane);

  v8f o0 = {}, o1 = {}, o2 = {}, o3 = {};
  float run_max = -1e30f, run_sum = 0.0f;

  for (int m0 = 0; m0 < 512; m0 += 32) {
    // St for m-chunks [m0,m0+16) and [m0+16,m0+32): rows = m, cols = n.
    v8f s0 = {}, s1 = {};
    {
      v16h a0 = load_tileA_h(k, 1024, m0, g * 64,      lane);
      v16h a1 = load_tileA_h(k, 1024, m0, g * 64 + 32, lane);
      s0 = wmma_f16(a0, bq0, s0);
      s0 = wmma_f16(a1, bq1, s0);
      a0 = load_tileA_h(k, 1024, m0 + 16, g * 64,      lane);
      a1 = load_tileA_h(k, 1024, m0 + 16, g * 64 + 32, lane);
      s1 = wmma_f16(a0, bq0, s1);
      s1 = wmma_f16(a1, bq1, s1);
    }

    // scale + positional log2-bias (lane owns n = n0+c; VGPR j owns m).
    const _Float16* lw = logw + ((size_t)(n0 + c) * 16 + g) * 512 + m0 + hs * 8;
    const h8 wlo = *(const h8*)(lw);
    const h8 whi = *(const h8*)(lw + 16);
#pragma unroll
    for (int j = 0; j < 8; ++j) {
      s0[j] = s0[j] * kScale + (float)wlo[j];
      s1[j] = s1[j] * kScale + (float)whi[j];
    }

    // online softmax (base 2): reduce over m (8 VGPRs + opposite lane-half).
    float tmax = s0[0];
#pragma unroll
    for (int j = 1; j < 8; ++j) tmax = fmaxf(tmax, s0[j]);
#pragma unroll
    for (int j = 0; j < 8; ++j) tmax = fmaxf(tmax, s1[j]);
    tmax = fmaxf(tmax, __shfl_xor(tmax, 16, 32));

    const float nmax  = fmaxf(run_max, tmax);
    const float alpha = fast_exp2(run_max - nmax);
    float psum = 0.0f;
#pragma unroll
    for (int j = 0; j < 8; ++j) {
      s0[j] = fast_exp2(s0[j] - nmax);
      s1[j] = fast_exp2(s1[j] - nmax);
      psum += s0[j] + s1[j];
    }
    psum += __shfl_xor(psum, 16, 32);
    run_sum = run_sum * alpha + psum;
    run_max = nmax;

    // rescale accumulator rows (row n = j+8*hs lives at lane j+8*hs).
#pragma unroll
    for (int j = 0; j < 8; ++j) {
      const float rs = __shfl(alpha, j + hs * 8, 32);
      o0[j] *= rs; o1[j] *= rs; o2[j] *= rs; o3[j] *= rs;
    }

    // lane-local repack of P into A-layout (16 n-rows x 32 m-K).
    v16h pa;
#pragma unroll
    for (int j = 0; j < 8; ++j) {
      pa[j]     = (_Float16)s0[j];
      pa[j + 8] = (_Float16)s1[j];
    }

    // O += P @ Vg  (VgT rows = output cols, contiguous in m).
    o0 = wmma_f16(pa, load_tileB_h(vgt, 512, g * 64,      m0, lane), o0);
    o1 = wmma_f16(pa, load_tileB_h(vgt, 512, g * 64 + 16, m0, lane), o1);
    o2 = wmma_f16(pa, load_tileB_h(vgt, 512, g * 64 + 32, m0, lane), o2);
    o3 = wmma_f16(pa, load_tileB_h(vgt, 512, g * 64 + 48, m0, lane), o3);
  }

  // epilogue: 1/rowsum and conv bias (f32 output).
  const float inv = 1.0f / run_sum;
  const float cb0 = conv_b[g * 64 + c];
  const float cb1 = conv_b[g * 64 + 16 + c];
  const float cb2 = conv_b[g * 64 + 32 + c];
  const float cb3 = conv_b[g * 64 + 48 + c];
#pragma unroll
  for (int j = 0; j < 8; ++j) {
    const float ri = __shfl(inv, j + hs * 8, 32);
    const size_t base = (size_t)(n0 + j + 8 * hs) * 1024 + g * 64 + c;
    out[base]      = o0[j] * ri + cb0;
    out[base + 16] = o1[j] * ri + cb1;
    out[base + 32] = o2[j] * ri + cb2;
    out[base + 48] = o3[j] * ri + cb3;
  }
}

// ---------------------------------------------------------------------------
// Launcher.  Workspace layout (f16, 28 MB total):
//   [ 0, 2MB)  roi_h  (1024x1024)     [ 2, 4MB)  qw_h  (1024x1024)
//   [ 4, 6MB)  kw_h   (1024x1024)     [ 6, 8MB)  cw_h  (1024x1024)
//   [ 8,10MB)  q_h    (1024x1024)     [10,11MB)  k_h   ( 512x1024)
//   [11,12MB)  vgt_h  (1024x 512)     [12,28MB)  logw  (1024x16x512)
// ---------------------------------------------------------------------------
extern "C" void kernel_launch(void* const* d_in, const int* in_sizes, int n_in,
                              void* d_out, int out_size, void* d_ws, size_t ws_size,
                              hipStream_t stream) {
  const float* roi    = (const float*)d_in[0];
  const float* pe     = (const float*)d_in[1];
  const float* pos_w  = (const float*)d_in[2];
  const float* pos_b  = (const float*)d_in[3];
  const float* q_w    = (const float*)d_in[4];
  const float* q_b    = (const float*)d_in[5];
  const float* k_w    = (const float*)d_in[6];
  const float* k_b    = (const float*)d_in[7];
  const float* conv_w = (const float*)d_in[8];
  const float* conv_b = (const float*)d_in[9];
  float* out = (float*)d_out;

  char* ws = (char*)d_ws;
  _Float16* roi_h  = (_Float16*)(ws);
  _Float16* qw_h   = (_Float16*)(ws + (size_t)( 2u << 20));
  _Float16* kw_h   = (_Float16*)(ws + (size_t)( 4u << 20));
  _Float16* cw_h   = (_Float16*)(ws + (size_t)( 6u << 20));
  _Float16* q_h    = (_Float16*)(ws + (size_t)( 8u << 20));
  _Float16* k_h    = (_Float16*)(ws + (size_t)(10u << 20));
  _Float16* vgt_h  = (_Float16*)(ws + (size_t)(11u << 20));
  _Float16* logw_h = (_Float16*)(ws + (size_t)(12u << 20));

  const int n1m = 1024 * 1024;
  // 0) pack f32 -> f16 operands (one shot, L2 resident).
  pack_f16_kernel<<<dim3(n1m / 8 / 256), 256, 0, stream>>>(roi,    roi_h, n1m);
  pack_f16_kernel<<<dim3(n1m / 8 / 256), 256, 0, stream>>>(q_w,    qw_h,  n1m);
  pack_f16_kernel<<<dim3(n1m / 8 / 256), 256, 0, stream>>>(k_w,    kw_h,  n1m);
  pack_f16_kernel<<<dim3(n1m / 8 / 256), 256, 0, stream>>>(conv_w, cw_h,  n1m);

  // 1) positional log2-bias (134MB NT stream): 524288 rows / 64 per block.
  posbias_kernel<<<dim3(8192), 128, 0, stream>>>(pe, pos_w, pos_b, logw_h);

  // 2) q = roi @ q_w^T + q_b            (M=1024, N=1024, K=1024)
  gemm_abt_f16_kernel<<<dim3(16, 16), 128, 0, stream>>>(roi_h, qw_h, q_b, q_h, 1024, 1024);

  // 3) k = roi[:512] @ k_w^T + k_b      (M=512, N=1024, K=1024)
  gemm_abt_f16_kernel<<<dim3(16, 8), 128, 0, stream>>>(roi_h, kw_h, k_b, k_h, 1024, 1024);

  // 4) VgT = conv_w @ roi[:512]^T       (M=1024, N=512, K=1024)
  gemm_abt_f16_kernel<<<dim3(8, 16), 128, 0, stream>>>(cw_h, roi_h, nullptr, vgt_h, 1024, 512);

  // 5) flash attention + fused output projection.
  flash_kernel<<<dim3(16, 16), 128, 0, stream>>>(q_h, k_h, vgt_h, logw_h, conv_b, out);
}
